// APPNPNet_27504970563789
// MI455X (gfx1250) — compile-verified, matrix operand order checked by reading, not run on hardware
//
#include <hip/hip_runtime.h>

// ---------------------------------------------------------------------------
// APPNP GNN on MI455X (gfx1250, wave32):
//   h = MLP(x) via v_wmma_f32_16x16x32_bf16, operands pre-swizzled in global
//   memory into the exact CDNA5 fragment layout (ISA 7.12.2), so the GEMM
//   inner loop is coalesced global_load_b128 -> wmma with no LDS staging.
//   Fragment double-buffering is parity-indexed (compile-time constants after
//   full unroll) so no cross-buffer register copies are generated.
//   Then 2 rounds of symmetric-normalized propagation (float atomics).
// ---------------------------------------------------------------------------

#define NNODES 100000          // multiple of 16 -> 6250 row tiles, no remainder
#define NEDGES 1600000
#define KDIM   256
#define KS     8               // KDIM / 32 k-steps
#define ALPHA_ 0.1f

typedef __attribute__((ext_vector_type(16))) __bf16 v16bf;
typedef __attribute__((ext_vector_type(8)))  float  v8f;

union FragBF {
    uint4 u4[2];   // 32 bytes = 16 bf16
    v16bf v;
};

__device__ __forceinline__ unsigned int f2bf_pack(float lo, float hi) {
    unsigned int a = __float_as_uint(lo);
    unsigned int b = __float_as_uint(hi);
    a = (a + 0x7FFFu + ((a >> 16) & 1u)) >> 16;   // round-to-nearest-even
    b = (b + 0x7FFFu + ((b >> 16) & 1u)) >> 16;
    return (b << 16) | (a & 0xFFFFu);
}

// K index held by fragment element e (0..15) for lane-group g (lane>>4),
// per CDNA5 ISA 7.12.2 (16-bit A 16x32 layout; B mirrors with M<->N).
__device__ __forceinline__ int kfrag(int g, int e) {
    return (e < 8) ? (g * 8 + e) : (16 + g * 8 + (e - 8));
}

__device__ __forceinline__ void loadFrag(FragBF& f, const unsigned int* p) {
    const uint4* q = (const uint4*)p;
    f.u4[0] = q[0];
    f.u4[1] = q[1];
}

// ---------------------------------------------------------------------------
// Swizzle a row-major fp32 activation [rowTiles*16 x 256] into bf16 fragment
// layout: entry index ((rt*KS + ks)*32 + lane) -> 16 contiguous halfs.
// ---------------------------------------------------------------------------
__global__ void swizzle_act(const float* __restrict__ A,
                            unsigned int* __restrict__ out, int rowTiles) {
    int t = blockIdx.x * blockDim.x + threadIdx.x;    // one fragment-lane entry
    if (t >= rowTiles * KS * 32) return;
    const int lane = t & 31;
    const int ks   = (t >> 5) & (KS - 1);
    const int rt   = t >> 8;
    const int r    = lane & 15;
    const int g    = lane >> 4;
    const float* ap = A + ((long)rt * 16 + r) * KDIM + ks * 32;
    float f[16];
#pragma unroll
    for (int j = 0; j < 8; ++j) { f[j] = ap[g * 8 + j]; f[8 + j] = ap[16 + g * 8 + j]; }
    unsigned int u[8];
#pragma unroll
    for (int i = 0; i < 8; ++i) u[i] = f2bf_pack(f[2 * i], f[2 * i + 1]);
    uint4* d = (uint4*)(out + (size_t)t * 8);
    uint4 q0; q0.x = u[0]; q0.y = u[1]; q0.z = u[2]; q0.w = u[3];
    uint4 q1; q1.x = u[4]; q1.y = u[5]; q1.z = u[6]; q1.w = u[7];
    d[0] = q0; d[1] = q1;
}

// ---------------------------------------------------------------------------
// Swizzle W [256 x NOUT] fp32 into bf16 B-fragment layout:
// packed-uint index ((ct*KS + ks)*32 + lane)*8 + e2.
// ---------------------------------------------------------------------------
__global__ void swizzle_W(const float* __restrict__ W,
                          unsigned int* __restrict__ out, int NOUT) {
    const int total = (NOUT / 16) * KS * 32 * 8;
    int idx = blockIdx.x * blockDim.x + threadIdx.x;
    if (idx >= total) return;
    const int e2    = idx & 7;
    const int entry = idx >> 3;
    const int lane  = entry & 31;
    const int ks    = (entry >> 5) & (KS - 1);
    const int ct    = entry >> 8;
    const int g     = lane >> 4;
    const int n     = ct * 16 + (lane & 15);
    const int k0    = ks * 32 + kfrag(g, 2 * e2);
    out[idx] = f2bf_pack(W[(long)k0 * NOUT + n], W[(long)(k0 + 1) * NOUT + n]);
}

// ---------------------------------------------------------------------------
// GEMM on pre-swizzled fragments: out = act(A[M x 256] * W[256 x NOUT] + b).
// 8 waves/block, one 16-row tile per wave, NOUT/16 f32 accumulators.
// Parity-indexed double buffering of A/B fragments; no LDS staging/barriers.
// OUTSWZ: emit bf16 fragment-swizzled output (feeds the next GEMM) via a
// per-wave LDS transpose slab (wave-private; LDS ops are in-order per wave).
// ---------------------------------------------------------------------------
template <int NOUT, bool RELU, bool OUTSWZ>
__global__ __launch_bounds__(256)
void gemm_frag(const unsigned int* __restrict__ Aswz,
               const unsigned int* __restrict__ Bswz,
               const float* __restrict__ bias,
               void* __restrict__ outp,
               int rowTiles) {
    constexpr int NT = NOUT / 16;                      // even (16 or 2)
    __shared__ float slab[OUTSWZ ? 8 * 16 * 32 : 8];   // 2 KB per wave when used

    const int tid  = threadIdx.x;
    const int lane = tid & 31;
    const int wave = tid >> 5;
    const int rt   = blockIdx.x * 8 + wave;
    if (rt >= rowTiles) return;                        // wave-uniform; EXEC stays full

    const unsigned int* aBase = Aswz + ((size_t)rt * KS * 32 + lane) * 8;
    const unsigned int* bBase = Bswz + (size_t)lane * 8;

    v8f acc[NT];
#pragma unroll
    for (int ct = 0; ct < NT; ++ct)
        acc[ct] = (v8f){0.f, 0.f, 0.f, 0.f, 0.f, 0.f, 0.f, 0.f};

    FragBF aF[2], bF[2];
    loadFrag(aF[0], aBase);
    loadFrag(bF[0], bBase);
#pragma unroll
    for (int ks = 0; ks < KS; ++ks) {
        if (ks + 1 < KS) loadFrag(aF[(ks + 1) & 1], aBase + (size_t)(ks + 1) * 256);
#pragma unroll
        for (int ct = 0; ct < NT; ++ct) {
            const int cur = ct & 1;                    // NT even -> parity of global idx
            const int nxt = cur ^ 1;
            if (ct + 1 < NT)
                loadFrag(bF[nxt], bBase + (size_t)((ct + 1) * KS + ks) * 256);
            else if (ks + 1 < KS)
                loadFrag(bF[nxt], bBase + (size_t)(ks + 1) * 256);
            acc[ct] = __builtin_amdgcn_wmma_f32_16x16x32_bf16(
                false, aF[ks & 1].v, false, bF[cur].v, (short)0, acc[ct], false, false);
        }
    }

    const int g  = lane >> 4;
    const int nl = lane & 15;

    if (OUTSWZ) {
        // transpose each 16x32 accumulator group through a wave-private slab,
        // then store packed bf16 fragments (32 contiguous bytes per lane).
        float* my = &slab[wave * 16 * 32];
        unsigned int* outw = (unsigned int*)outp;
#pragma unroll
        for (int t2 = 0; t2 < NT / 2; ++t2) {          // = ks of the consumer GEMM
#pragma unroll
            for (int half = 0; half < 2; ++half) {
                const int ct = 2 * t2 + half;
                const float bv = bias[ct * 16 + nl];
#pragma unroll
                for (int v = 0; v < 8; ++v) {
                    float val = acc[ct][v] + bv;
                    if (RELU) val = fmaxf(val, 0.f);
                    my[(v + 8 * g) * 32 + half * 16 + nl] = val;
                }
            }
            float f[16];
#pragma unroll
            for (int e = 0; e < 16; ++e) f[e] = my[(lane & 15) * 32 + kfrag(g, e)];
            unsigned int u[8];
#pragma unroll
            for (int i = 0; i < 8; ++i) u[i] = f2bf_pack(f[2 * i], f[2 * i + 1]);
            uint4* d = (uint4*)(outw + ((size_t)(rt * KS + t2) * 32 + lane) * 8);
            uint4 q0; q0.x = u[0]; q0.y = u[1]; q0.z = u[2]; q0.w = u[3];
            uint4 q1; q1.x = u[4]; q1.y = u[5]; q1.z = u[6]; q1.w = u[7];
            d[0] = q0; d[1] = q1;
        }
    } else {
        float* outf = (float*)outp;
#pragma unroll
        for (int ct = 0; ct < NT; ++ct) {
            const float bv = bias[ct * 16 + nl];
#pragma unroll
            for (int v = 0; v < 8; ++v) {
                const long row = (long)rt * 16 + 8 * g + v;
                float val = acc[ct][v] + bv;
                if (RELU) val = fmaxf(val, 0.f);
                outf[row * NOUT + ct * 16 + nl] = val;
            }
        }
    }
}

// ---------------------------------------------------------------------------
// Propagation kernels
// ---------------------------------------------------------------------------
__global__ void deg_init(float* deg, int n) {
    int i = blockIdx.x * blockDim.x + threadIdx.x;
    if (i < n) deg[i] = 1.0f;                          // self-loop
}

__global__ void deg_count(const long long* __restrict__ dst, float* deg, int e) {
    int i = blockIdx.x * blockDim.x + threadIdx.x;
    if (i < e) atomicAdd(&deg[dst[i]], 1.0f);
}

__global__ void dinv_calc(const float* __restrict__ deg, float* dinv, int n) {
    int i = blockIdx.x * blockDim.x + threadIdx.x;
    if (i < n) dinv[i] = rsqrtf(deg[i]);               // deg >= 1 always
}

// agg[i] = h[i] * dinv[i]^2   (self-loop term; also initializes agg)
__global__ void appnp_self(const float* __restrict__ h, const float* __restrict__ dinv,
                           float* __restrict__ agg, int n) {
    int t = blockIdx.x * blockDim.x + threadIdx.x;
    if (t >= n * 8) return;
    int i = t >> 3;
    int c = (t & 7) * 4;
    float s = dinv[i] * dinv[i];
    const float4 hv = *(const float4*)(h + (long)i * 32 + c);
    float4 o;
    o.x = hv.x * s; o.y = hv.y * s; o.z = hv.z * s; o.w = hv.w * s;
    *(float4*)(agg + (long)i * 32 + c) = o;
}

// agg[dst] += h[src] * dinv[src]*dinv[dst]   (8 threads/edge, float4 each)
__global__ void appnp_scatter(const long long* __restrict__ src,
                              const long long* __restrict__ dst,
                              const float* __restrict__ h,
                              const float* __restrict__ dinv,
                              float* __restrict__ agg, int e) {
    int t = blockIdx.x * blockDim.x + threadIdx.x;
    if (t >= e * 8) return;
    int  ei = t >> 3;
    int  c  = (t & 7) * 4;
    long long s = src[ei];
    long long d = dst[ei];
    float norm = dinv[s] * dinv[d];
    const float4 hv = *(const float4*)(h + s * 32 + c);
    float* o = agg + d * 32 + c;
    atomicAdd(o + 0, hv.x * norm);
    atomicAdd(o + 1, hv.y * norm);
    atomicAdd(o + 2, hv.z * norm);
    atomicAdd(o + 3, hv.w * norm);
}

// hnew = (1-alpha)*agg + alpha*h0
__global__ void appnp_update(const float* __restrict__ agg, const float* __restrict__ h0,
                             float* __restrict__ hnew, int n) {
    int t = blockIdx.x * blockDim.x + threadIdx.x;
    if (t < n * 32) hnew[t] = (1.0f - ALPHA_) * agg[t] + ALPHA_ * h0[t];
}

// ---------------------------------------------------------------------------
extern "C" void kernel_launch(void* const* d_in, const int* in_sizes, int n_in,
                              void* d_out, int out_size, void* d_ws, size_t ws_size,
                              hipStream_t stream) {
    const float*     x     = (const float*)d_in[0];
    const long long* ei    = (const long long*)d_in[1];   // [2, E] int64
    const float*     W_in  = (const float*)d_in[2];
    const float*     b_in  = (const float*)d_in[3];
    const float*     W_h   = (const float*)d_in[4];
    const float*     b_h   = (const float*)d_in[5];
    const float*     W_out = (const float*)d_in[6];
    const float*     b_out = (const float*)d_in[7];
    float*           out   = (float*)d_out;

    const long long* src = ei;
    const long long* dst = ei + NEDGES;

    const int rowTiles = NNODES / 16;                  // 6250 (exact)
    const size_t actSwzU = (size_t)rowTiles * KS * 32 * 8;   // uints per swizzled act
    const size_t w256U   = 16 * KS * 32 * 8;                 // 32768
    const size_t w32U    = 2 * KS * 32 * 8;                  // 4096

    // workspace layout (4-byte units)
    unsigned int* xswz   = (unsigned int*)d_ws;
    unsigned int* h1swz  = xswz  + actSwzU;
    unsigned int* h2swz  = h1swz + actSwzU;
    unsigned int* wInSwz = h2swz + actSwzU;
    unsigned int* wHSwz  = wInSwz + w256U;
    unsigned int* wOSwz  = wHSwz  + w256U;
    float* h0   = (float*)(wOSwz + w32U);              // [N,32]
    float* ht   = h0  + (size_t)NNODES * 32;
    float* agg  = ht  + (size_t)NNODES * 32;
    float* deg  = agg + (size_t)NNODES * 32;
    float* dinv = deg + NNODES;

    // --- pre-swizzle x and weights into WMMA fragment layouts ---
    swizzle_act<<<(rowTiles * KS * 32 + 255) / 256, 256, 0, stream>>>(x, xswz, rowTiles);
    swizzle_W<<<(int)((16 * KS * 32 * 8 + 255) / 256), 256, 0, stream>>>(W_in,  wInSwz, 256);
    swizzle_W<<<(int)((16 * KS * 32 * 8 + 255) / 256), 256, 0, stream>>>(W_h,   wHSwz,  256);
    swizzle_W<<<(int)((2  * KS * 32 * 8 + 255) / 256), 256, 0, stream>>>(W_out, wOSwz,  32);

    // --- MLP encoder: load-fragment -> wmma, swizzled bf16 chaining ---
    const int gemmGrid = (rowTiles + 7) / 8;           // 782
    gemm_frag<256, true,  true ><<<gemmGrid, 256, 0, stream>>>(xswz,  wInSwz, b_in,  h1swz, rowTiles);
    gemm_frag<256, true,  true ><<<gemmGrid, 256, 0, stream>>>(h1swz, wHSwz,  b_h,   h2swz, rowTiles);
    gemm_frag<32,  false, false><<<gemmGrid, 256, 0, stream>>>(h2swz, wOSwz,  b_out, h0,    rowTiles);

    // --- gcn_norm degrees (with self-loops) ---
    deg_init <<<(NNODES + 255) / 256, 256, 0, stream>>>(deg, NNODES);
    deg_count<<<(NEDGES + 255) / 256, 256, 0, stream>>>(dst, deg, NEDGES);
    dinv_calc<<<(NNODES + 255) / 256, 256, 0, stream>>>(deg, dinv, NNODES);

    const int selfGrid = (NNODES * 8 + 255) / 256;
    const int scatGrid = (int)(((long long)NEDGES * 8 + 255) / 256);
    const int updGrid  = (NNODES * 32 + 255) / 256;

    // iteration 1: h = h0
    appnp_self   <<<selfGrid, 256, 0, stream>>>(h0, dinv, agg, NNODES);
    appnp_scatter<<<scatGrid, 256, 0, stream>>>(src, dst, h0, dinv, agg, NEDGES);
    appnp_update <<<updGrid, 256, 0, stream>>>(agg, h0, ht, NNODES);

    // iteration 2: h = ht, write final result to d_out
    appnp_self   <<<selfGrid, 256, 0, stream>>>(ht, dinv, agg, NNODES);
    appnp_scatter<<<scatGrid, 256, 0, stream>>>(src, dst, ht, dinv, agg, NEDGES);
    appnp_update <<<updGrid, 256, 0, stream>>>(agg, h0, out, NNODES);
}
